// TorchElementSpecificA2MDNN_72378788872232
// MI455X (gfx1250) — matrix operand order repaired and checked by reading, not compiled, not run on hardware
//
#include <hip/hip_runtime.h>
#include <hip/hip_bf16.h>

// ---------------------------------------------------------------------------
// MoE expert MLP (384->128->96->2, celu, 4 experts) on gfx1250.
// Strategy: f16 WMMA (v_wmma_f32_16x16x32_f16) with fp32 accumulation.
//   - features (201 MB fp32) read from HBM exactly ONCE, staged to LDS as f16
//   - per-expert weights pre-transposed/converted to f16 in d_ws by a prep
//     kernel, staged into LDS per expert phase (L2-resident, cheap)
//   - dense 4-expert compute per 16-token tile, per-row select on store
// LDS: 265 KB/WG (legal on CDNA5: up to 320 KB/WG), dynamic alloc.
// ---------------------------------------------------------------------------

typedef _Float16 v16h __attribute__((ext_vector_type(16)));
typedef _Float16 v8h  __attribute__((ext_vector_type(8)));
typedef _Float16 v4h  __attribute__((ext_vector_type(4)));
typedef float    v8f  __attribute__((ext_vector_type(8)));
typedef float    v4f  __attribute__((ext_vector_type(4)));

#define T_TOK   131072      // 64 * 2048 tokens
#define NTILES  8192        // T_TOK / 16
#define F_IN    384
#define H1N     128
#define H2N     96
#define N_EXP   4

// LDS pitches in halves (padded to avoid bank conflicts)
#define XP   392
#define W1P  392
#define W2P  136
#define W3P  104
#define HP   136

// d_ws layout (in halves): transposed f16 weights [e][n][k]
#define W1T_OFF   0         // 4 * 128 * 384
#define W2T_OFF   196608    // 4 * 96  * 128
#define W3T_OFF   245760    // 4 * 16  * 96  (cols 2..15 zero-padded)
#define WST_TOTAL 251904    // halves -> 503808 bytes needed in d_ws

// shared memory byte offsets
#define SM_W1    0
#define SM_W2    100352     // 128*392*2
#define SM_W3    126464     // + 96*136*2
#define SM_H1    129792     // + 16*104*2
#define SM_X     164608     // + 8 waves * 16*136*2
#define SM_BYTES 264960     // + 8 waves * 16*392*2

__device__ __forceinline__ v8f wmma_f16(v16h a, v16h b, v8f c) {
  return __builtin_amdgcn_wmma_f32_16x16x32_f16(false, a, false, b, (short)0, c,
                                                false, false);
}

// A fragment (16x32 f16): lane = 16*h + nl holds row nl,
// halves 0..7  = K (kk*32 + h*8 + 0..7), halves 8..15 = K (+16).
__device__ __forceinline__ v16h frag_a(const _Float16* buf, int pitch, int kk,
                                       int nl, int h) {
  const _Float16* p = buf + nl * pitch + kk * 32 + h * 8;
  v8h lo = *(const v8h*)p;
  v8h hi = *(const v8h*)(p + 16);
  return __builtin_shufflevector(lo, hi, 0, 1, 2, 3, 4, 5, 6, 7, 8, 9, 10, 11,
                                 12, 13, 14, 15);
}

// B fragment (32x16 f16): lane = 16*h + nl holds column (nn*16+nl),
// halves 0..7 = K (kk*32 + h*8 + 0..7), halves 8..15 = K (+16).  buf is [col][k] (W transposed).
__device__ __forceinline__ v16h frag_b(const _Float16* buf, int pitch, int kk,
                                       int nn, int nl, int h) {
  const _Float16* p = buf + (nn * 16 + nl) * pitch + kk * 32 + h * 8;
  v8h lo = *(const v8h*)p;
  v8h hi = *(const v8h*)(p + 16);
  return __builtin_shufflevector(lo, hi, 0, 1, 2, 3, 4, 5, 6, 7, 8, 9, 10, 11,
                                 12, 13, 14, 15);
}

__device__ __forceinline__ float celu1(float x) {
  return x > 0.0f ? x : (__expf(x) - 1.0f);
}

// ---------------------------------------------------------------------------
// Prep: convert + transpose all expert weights to f16 in d_ws.
// ---------------------------------------------------------------------------
__global__ void prep_weights(const float* __restrict__ W1,
                             const float* __restrict__ W2,
                             const float* __restrict__ W3,
                             _Float16* __restrict__ wt) {
  int idx = blockIdx.x * blockDim.x + threadIdx.x;
  if (idx < W2T_OFF) {                       // W1t[e][n][k], 4*128*384
    int e = idx / (F_IN * H1N), r = idx % (F_IN * H1N);
    int n = r / F_IN, k = r % F_IN;
    wt[idx] = (_Float16)W1[((size_t)e * F_IN + k) * H1N + n];
  } else if (idx < W3T_OFF) {                // W2t[e][n][k], 4*96*128
    int j = idx - W2T_OFF;
    int e = j / (H1N * H2N), r = j % (H1N * H2N);
    int n = r / H1N, k = r % H1N;
    wt[idx] = (_Float16)W2[((size_t)e * H1N + k) * H2N + n];
  } else if (idx < WST_TOTAL) {              // W3t[e][n][k], 4*16*96 (n>=2 -> 0)
    int j = idx - W3T_OFF;
    int e = j / (16 * H2N), r = j % (16 * H2N);
    int n = r / H2N, k = r % H2N;
    wt[idx] = (n < 2) ? (_Float16)W3[((size_t)e * H2N + k) * 2 + n]
                      : (_Float16)0.0f;
  }
}

// ---------------------------------------------------------------------------
// Main: 256 threads = 8 waves, one 16-token tile per wave per grid step.
// ---------------------------------------------------------------------------
__global__ __launch_bounds__(256) void moe_main(
    const int* __restrict__ elements, const float* __restrict__ X,
    const _Float16* __restrict__ wt, const float* __restrict__ b1,
    const float* __restrict__ b2, const float* __restrict__ b3,
    float* __restrict__ out) {
  extern __shared__ __align__(16) char smem[];
  _Float16* w1 = (_Float16*)(smem + SM_W1);
  _Float16* w2 = (_Float16*)(smem + SM_W2);
  _Float16* w3 = (_Float16*)(smem + SM_W3);

  const int tid = threadIdx.x;
  const int wave = tid >> 5, lane = tid & 31;
  const int nl = lane & 15, h = lane >> 4;
  _Float16* hb = (_Float16*)(smem + SM_H1) + wave * (16 * HP);
  _Float16* xb = (_Float16*)(smem + SM_X) + wave * (16 * XP);

  const v8f vzero = {0.f, 0.f, 0.f, 0.f, 0.f, 0.f, 0.f, 0.f};

  for (int base = blockIdx.x * 8; base < NTILES; base += gridDim.x * 8) {
    const int tile = base + wave;
    const bool valid = tile < NTILES;

    if (valid) {
      // Stage this wave's 16x384 fp32 tile -> f16 LDS (nontemporal reads:
      // X is a 201MB single-use stream; keep L2 for weights).
      const float* xs = X + (size_t)tile * (16 * F_IN);
      for (int j = 0; j < 48; ++j) {
        int c = lane + 32 * j;
        int r = c / 96, k4 = (c % 96) * 4;
        v4f f = __builtin_nontemporal_load((const v4f*)(xs + r * F_IN + k4));
        v4h hh;
        hh[0] = (_Float16)f[0]; hh[1] = (_Float16)f[1];
        hh[2] = (_Float16)f[2]; hh[3] = (_Float16)f[3];
        *(v4h*)(xb + r * XP + k4) = hh;
      }
      // Output #1: elements passthrough (numeric).
      if (lane < 16) {
        int t = tile * 16 + lane;
        out[t] = (float)elements[t];
      }
    }

    for (int e = 0; e < N_EXP; ++e) {
      __syncthreads();  // previous phase compute done before weight overwrite
      // Stage expert e's f16 weights into LDS (coalesced b128 copies).
      {
        const _Float16* s1 = wt + W1T_OFF + e * (F_IN * H1N);
        #pragma unroll
        for (int j = 0; j < 24; ++j) {
          int c = tid + 256 * j;
          int n = c / 48, kc = (c % 48) * 8;
          *(v8h*)(w1 + n * W1P + kc) = *(const v8h*)(s1 + n * F_IN + kc);
        }
        const _Float16* s2 = wt + W2T_OFF + e * (H1N * H2N);
        #pragma unroll
        for (int j = 0; j < 6; ++j) {
          int c = tid + 256 * j;
          int n = c / 16, kc = (c % 16) * 8;
          *(v8h*)(w2 + n * W2P + kc) = *(const v8h*)(s2 + n * H1N + kc);
        }
        if (tid < 192) {
          int n = tid / 12, kc = (tid % 12) * 8;
          *(v8h*)(w3 + n * W3P + kc) =
              *(const v8h*)(wt + W3T_OFF + e * (16 * H2N) + n * H2N + kc);
        }
      }
      __syncthreads();
      if (!valid) continue;

      // ---- Layer 1: H1[16x128] = X[16x384] @ W1 ------------------------
      v8f acc1[8];
      #pragma unroll
      for (int nn = 0; nn < 8; ++nn) acc1[nn] = vzero;
      for (int kk = 0; kk < F_IN / 32; ++kk) {
        v16h a = frag_a(xb, XP, kk, nl, h);
        #pragma unroll
        for (int nn = 0; nn < 8; ++nn)
          acc1[nn] = wmma_f16(a, frag_b(w1, W1P, kk, nn, nl, h), acc1[nn]);
      }
      // bias + celu -> f16 H1 in LDS (C/D layout: row v+8h, col nn*16+nl)
      #pragma unroll
      for (int nn = 0; nn < 8; ++nn) {
        float bi = b1[e * H1N + nn * 16 + nl];
        #pragma unroll
        for (int v = 0; v < 8; ++v)
          hb[(v + 8 * h) * HP + nn * 16 + nl] =
              (_Float16)celu1(acc1[nn][v] + bi);
      }

      // ---- Layer 2: H2[16x96] = H1[16x128] @ W2 ------------------------
      v8f acc2[6];
      #pragma unroll
      for (int nn = 0; nn < 6; ++nn) acc2[nn] = vzero;
      for (int kk = 0; kk < H1N / 32; ++kk) {
        v16h a = frag_a(hb, HP, kk, nl, h);
        #pragma unroll
        for (int nn = 0; nn < 6; ++nn)
          acc2[nn] = wmma_f16(a, frag_b(w2, W2P, kk, nn, nl, h), acc2[nn]);
      }
      #pragma unroll
      for (int nn = 0; nn < 6; ++nn) {   // reuse hb for H2 (reads are done)
        float bi = b2[e * H2N + nn * 16 + nl];
        #pragma unroll
        for (int v = 0; v < 8; ++v)
          hb[(v + 8 * h) * HP + nn * 16 + nl] =
              (_Float16)celu1(acc2[nn][v] + bi);
      }

      // ---- Layer 3: O[16x2] = H2[16x96] @ W3 (N zero-padded to 16) -----
      v8f acc3 = vzero;
      for (int kk = 0; kk < H2N / 32; ++kk) {
        v16h a = frag_a(hb, HP, kk, nl, h);
        acc3 = wmma_f16(a, frag_b(w3, W3P, kk, 0, nl, h), acc3);
      }
      // Per-row expert select + store (output #2 at offset T_TOK).
      if (nl < 2) {
        float bi = b3[e * 2 + nl];
        #pragma unroll
        for (int v = 0; v < 8; ++v) {
          int t = tile * 16 + v + 8 * h;
          if (elements[t] == e) out[(size_t)T_TOK + t * 2 + nl] = acc3[v] + bi;
        }
      }
    }
  }
}

extern "C" void kernel_launch(void* const* d_in, const int* in_sizes, int n_in,
                              void* d_out, int out_size, void* d_ws,
                              size_t ws_size, hipStream_t stream) {
  const int*   elements = (const int*)d_in[0];
  const float* X        = (const float*)d_in[1];
  const float* W1       = (const float*)d_in[2];
  const float* b1       = (const float*)d_in[3];
  const float* W2       = (const float*)d_in[4];
  const float* b2       = (const float*)d_in[5];
  const float* W3       = (const float*)d_in[6];
  const float* b3       = (const float*)d_in[7];
  float*       out      = (float*)d_out;
  (void)in_sizes; (void)n_in; (void)out_size; (void)ws_size;

  // Allow 265 KB dynamic LDS (CDNA5 supports up to 320 KB per workgroup).
  hipFuncSetAttribute((const void*)moe_main,
                      hipFuncAttributeMaxDynamicSharedMemorySize, SM_BYTES);

  _Float16* wt = (_Float16*)d_ws;  // needs 503,808 bytes of scratch
  prep_weights<<<(WST_TOTAL + 255) / 256, 256, 0, stream>>>(W1, W2, W3, wt);
  moe_main<<<NTILES / 8, 256, SM_BYTES, stream>>>(elements, X, wt, b1, b2, b3,
                                                  out);
}
